// LLaDA2HybridLinearSoftmaxFLAWrapper_4784593567893
// MI455X (gfx1250) — matrix operand order repaired and checked by reading, not compiled, not run on hardware
//
#include <hip/hip_runtime.h>

// ---------------------------------------------------------------------------
// MI455X (gfx1250) implementation.
// ~275 GFLOP of dense GEMM (bf16 WMMA, f32 accum, pre-transposed weights,
// async global->LDS double-buffered pipeline) + ~21 GFLOP of attention/scan
// math (f32 WMMA 16x16x4, reference precision). ~180MB traffic -> compute
// bound at 23.3 TB/s.
// ---------------------------------------------------------------------------

typedef __attribute__((ext_vector_type(16))) __bf16 v16bf;
typedef __attribute__((ext_vector_type(8)))  __bf16 v8bf;
typedef __attribute__((ext_vector_type(8)))  float  v8f;
typedef __attribute__((ext_vector_type(2)))  float  v2f;

#define BB   2
#define TT   4096
#define HID  2048
#define HH   16
#define KK_  128
#define BS   64
#define NB   64
#define QKVN 6144   // fused q|k|v columns

__device__ __constant__ const float kScaling = 0.08838834764831843f; // 128^-0.5
#define EPS_ 1e-6f

// ---------------- CDNA5 async global->LDS (ASYNCcnt-tracked) ---------------

__device__ __forceinline__ void async_ld_b128(const void* g, const void* l) {
  // GLOBAL_LOAD_ASYNC_TO_LDS_B128: VDST = per-lane LDS byte offset, VADDR = 64b
  // global address. Low 32 bits of a flat LDS pointer are the LDS offset.
  asm volatile("global_load_async_to_lds_b128 %0, %1, off"
               :: "v"((unsigned)(unsigned long long)l), "v"(g)
               : "memory");
}
#define WAIT_ASYNC_0() asm volatile("s_wait_asynccnt 0x0" ::: "memory")
#define WAIT_ASYNC_8() asm volatile("s_wait_asynccnt 0x8" ::: "memory")

// ---------------- WMMA fragment helpers (wave32 layouts per ISA 7.12.2) ----

// 16-bit A 16x32: lane g=lane>>4 holds K = {g*8+0..7, g*8+16..23}, row = m0+(lane&15)
__device__ __forceinline__ v16bf load_a_bf16(const __bf16* sm, int ld, int m0, int lane) {
  int g = lane >> 4;
  const __bf16* r = sm + (m0 + (lane & 15)) * ld;
  v8bf lo = *(const v8bf*)(r + g * 8);
  v8bf hi = *(const v8bf*)(r + g * 8 + 16);
  v16bf o;
#pragma unroll
  for (int i = 0; i < 8; ++i) { o[i] = lo[i]; o[i + 8] = hi[i]; }
  return o;
}

// 16-bit B 32x16 from LDS stored column-major [n][k]: lane holds col n0+(lane&15),
// K = {g*8+0..7, g*8+16..23}.
__device__ __forceinline__ v16bf load_b_bf16(const __bf16* smT, int ld, int n0, int lane) {
  int g = lane >> 4;
  const __bf16* c = smT + (n0 + (lane & 15)) * ld;
  v8bf lo = *(const v8bf*)(c + g * 8);
  v8bf hi = *(const v8bf*)(c + g * 8 + 16);
  v16bf o;
#pragma unroll
  for (int i = 0; i < 8; ++i) { o[i] = lo[i]; o[i + 8] = hi[i]; }
  return o;
}

// f32 A 16x4: VGPR i -> K = kk + g*2 + i, row = m0+(lane&15)
__device__ __forceinline__ v2f load_f32_a(const float* Am, int ld, int m0, int kk, int lane) {
  int g = lane >> 4;
  const float* p = Am + (m0 + (lane & 15)) * ld + kk + g * 2;
  v2f a; a[0] = p[0]; a[1] = p[1];
  return a;
}
// f32 B 4x16 (B row-major [k][n]): VGPR i -> row kk+g*2+i, col n0+(lane&15)
__device__ __forceinline__ v2f load_f32_b(const float* Bm, int ld, int n0, int kk, int lane) {
  int g = lane >> 4, col = n0 + (lane & 15);
  v2f b; b[0] = Bm[(kk + g * 2) * ld + col]; b[1] = Bm[(kk + g * 2 + 1) * ld + col];
  return b;
}
// f32 B where source is stored [n][k] (i.e. B[k][n] = Bm[n*ld + k])
__device__ __forceinline__ v2f load_f32_bT(const float* Bm, int ld, int n0, int kk, int lane) {
  int g = lane >> 4;
  const float* p = Bm + (n0 + (lane & 15)) * ld + kk + g * 2;
  v2f b; b[0] = p[0]; b[1] = p[1];
  return b;
}
// C/D 16x16 f32: VGPR r -> row m0 + r + 8*g, col n0+(lane&15)
__device__ __forceinline__ v8f load_cd(const float* m, int ld, int m0, int n0, int lane) {
  int g = lane >> 4, col = n0 + (lane & 15);
  v8f c;
#pragma unroll
  for (int r = 0; r < 8; ++r) c[r] = m[(m0 + r + 8 * g) * ld + col];
  return c;
}
__device__ __forceinline__ void store_cd(float* m, int ld, int m0, int n0, int lane, v8f c) {
  int g = lane >> 4, col = n0 + (lane & 15);
#pragma unroll
  for (int r = 0; r < 8; ++r) m[(m0 + r + 8 * g) * ld + col] = c[r];
}

__device__ __forceinline__ float phi_f(float z, float ps, float pb) {
  float w = z * ps + pb;
  float e = (w > 0.f) ? (w + 1.f) : __expf(w);   // elu(w)+1
  return fminf(fmaxf(e, 0.f), 10.f);
}

// ---------------- conversion kernels ---------------------------------------

__global__ void cvt_f32_bf16(const float* __restrict__ in, __bf16* __restrict__ out, size_t n) {
  size_t i = (size_t)blockIdx.x * blockDim.x + threadIdx.x;
  if (i < n) out[i] = (__bf16)in[i];
}

// W: [Kd][cols] row-major f32 -> out: bf16 transposed, column (off+c) stored as
// a contiguous row of length Kd: out[(off+c)*Kd + r] = W[r][c]
__global__ void pack_wT_bf16(const float* __restrict__ W, __bf16* __restrict__ out,
                             int Kd, int cols, int off) {
  size_t i = (size_t)blockIdx.x * blockDim.x + threadIdx.x;
  if (i >= (size_t)Kd * cols) return;
  int r = (int)(i / cols), c = (int)(i % cols);
  out[(size_t)(off + c) * Kd + r] = (__bf16)W[i];
}

// ---------------- bf16 GEMM: C[M,N] = A[M,Kd] * Bt[N,Kd]^T (+bias) ---------
// Block 256 threads = 8 waves (4Mx2N); macro tile 128x128; per-wave 32x64
// (2 A-frags x 4 B-frags = 8 wmma per 32-K step); TK=64 -> 16 wmma between
// barrier pairs. LDS tiles are filled with GLOBAL_LOAD_ASYNC_TO_LDS_B128,
// double buffered: next tile's fill overlaps current tile's WMMA.

#define GTM 128
#define GTN 128
#define GTK 64

__global__ __launch_bounds__(256) void gemm_bf16_rcr(
    const __bf16* __restrict__ A, const __bf16* __restrict__ Bt,
    const float* __restrict__ bias, float* __restrict__ C,
    int M, int N, int Kd) {
  __shared__ __align__(16) __bf16 As[2][GTM * GTK];   // [m][k]
  __shared__ __align__(16) __bf16 Bs[2][GTN * GTK];   // [n][k]

  int tid = threadIdx.x, lane = tid & 31, wave = tid >> 5;
  int wm = wave >> 1, wn = wave & 1;               // 4 x 2 wave grid
  int bm = blockIdx.y * GTM, bn = blockIdx.x * GTN;

  // tile-fill mapping: 128 rows x 64 cols, 2 threads per row, 32 bf16 each
  int lrow = tid >> 1, lhalf = (tid & 1) * 32;
  const __bf16* srcA = A  + (size_t)(bm + lrow) * Kd + lhalf;
  const __bf16* srcB = Bt + (size_t)(bn + lrow) * Kd + lhalf;
  int ldst = lrow * GTK + lhalf;

  v8f acc[2][4];
#pragma unroll
  for (int mi = 0; mi < 2; ++mi)
#pragma unroll
    for (int ni = 0; ni < 4; ++ni) acc[mi][ni] = (v8f){};

  // prologue: fill buffer 0 (8 async b128 per thread)
#pragma unroll
  for (int j = 0; j < 4; ++j) {
    async_ld_b128(srcA + j * 8, &As[0][ldst + j * 8]);
    async_ld_b128(srcB + j * 8, &Bs[0][ldst + j * 8]);
  }

  int cur = 0;
  for (int k0 = 0; k0 < Kd; k0 += GTK) {
    bool more = (k0 + GTK) < Kd;
    if (more) {   // start next tile into the other buffer
#pragma unroll
      for (int j = 0; j < 4; ++j) {
        async_ld_b128(srcA + k0 + GTK + j * 8, &As[cur ^ 1][ldst + j * 8]);
        async_ld_b128(srcB + k0 + GTK + j * 8, &Bs[cur ^ 1][ldst + j * 8]);
      }
      WAIT_ASYNC_8();   // async done in order: cnt<=8 => current tile landed
    } else {
      WAIT_ASYNC_0();
    }
    __syncthreads();

#pragma unroll
    for (int kc = 0; kc < GTK; kc += 32) {
      v16bf afr[2], bfr[4];
#pragma unroll
      for (int mi = 0; mi < 2; ++mi)
        afr[mi] = load_a_bf16(&As[cur][kc], GTK, wm * 32 + mi * 16, lane);
#pragma unroll
      for (int ni = 0; ni < 4; ++ni)
        bfr[ni] = load_b_bf16(&Bs[cur][kc], GTK, wn * 64 + ni * 16, lane);
#pragma unroll
      for (int mi = 0; mi < 2; ++mi)
#pragma unroll
        for (int ni = 0; ni < 4; ++ni)
          acc[mi][ni] = __builtin_amdgcn_wmma_f32_16x16x32_bf16(
              false, afr[mi], false, bfr[ni], (short)0, acc[mi][ni], false, false);
    }
    __syncthreads();   // all waves done reading buf cur before it is refilled
    cur ^= 1;
  }

  int g = lane >> 4, lc = lane & 15;
#pragma unroll
  for (int mi = 0; mi < 2; ++mi) {
    int rowb = bm + wm * 32 + mi * 16 + 8 * g;
#pragma unroll
    for (int ni = 0; ni < 4; ++ni) {
      int n = bn + wn * 64 + ni * 16 + lc;
      float bv = bias ? bias[n] : 0.0f;
#pragma unroll
      for (int r = 0; r < 8; ++r)
        C[(size_t)(rowb + r) * N + n] = acc[mi][ni][r] + bv;
    }
  }
}

// ---------------- local block attention (writes hyb = local) ---------------

__global__ __launch_bounds__(256) void local_attn_kernel(
    const float* __restrict__ qkv, float* __restrict__ hyb) {
  __shared__ __align__(16) float Qs[BS * KK_];
  __shared__ __align__(16) float Ks[BS * KK_];
  __shared__ __align__(16) float Vs[BS * KK_];
  __shared__ __align__(16) float Ss[BS * BS];

  int tid = threadIdx.x, lane = tid & 31, wave = tid >> 5;
  int blk = blockIdx.x;
  int nbk = blk & 63, h = (blk >> 6) & 15, b = blk >> 10;
  int t0 = nbk * BS;
  size_t rowbase = (size_t)b * TT;

  for (int i = tid; i < BS * KK_; i += 256) {
    int t = i >> 7, k = i & 127;
    size_t base = (rowbase + t0 + t) * (size_t)QKVN + h * KK_ + k;
    Qs[i] = qkv[base] * kScaling;
    Ks[i] = qkv[base + 2048];
    Vs[i] = qkv[base + 4096];
  }
  __syncthreads();

  // scores = Q K^T (64x64, contract 128)
  for (int tile = wave; tile < 16; tile += 8) {
    int m0 = (tile >> 2) * 16, n0 = (tile & 3) * 16;
    v8f acc = {};
    for (int kk = 0; kk < KK_; kk += 4) {
      v2f a = load_f32_a(Qs, KK_, m0, kk, lane);
      v2f bb = load_f32_bT(Ks, KK_, n0, kk, lane);   // B[k][s] = Ks[s][k]
      acc = __builtin_amdgcn_wmma_f32_16x16x4_f32(false, a, false, bb,
                                                  (short)0, acc, false, false);
    }
    store_cd(Ss, BS, m0, n0, lane, acc);
  }
  __syncthreads();

  if (tid < BS) { // row softmax, f32
    float mx = -3.0e38f;
    for (int s = 0; s < BS; ++s) mx = fmaxf(mx, Ss[tid * BS + s]);
    float sum = 0.f;
    for (int s = 0; s < BS; ++s) { float e = __expf(Ss[tid * BS + s] - mx); Ss[tid * BS + s] = e; sum += e; }
    float inv = 1.f / sum;
    for (int s = 0; s < BS; ++s) Ss[tid * BS + s] *= inv;
  }
  __syncthreads();

  // out = attn V (64x128, contract 64) -> hyb
  for (int tile = wave; tile < 32; tile += 8) {
    int m0 = (tile >> 3) * 16, n0 = (tile & 7) * 16;
    v8f acc = {};
    for (int kk = 0; kk < BS; kk += 4) {
      v2f a = load_f32_a(Ss, BS, m0, kk, lane);
      v2f bb = load_f32_b(Vs, KK_, n0, kk, lane);
      acc = __builtin_amdgcn_wmma_f32_16x16x4_f32(false, a, false, bb,
                                                  (short)0, acc, false, false);
    }
    int g = lane >> 4, col = n0 + (lane & 15);
#pragma unroll
    for (int r = 0; r < 8; ++r)
      hyb[(rowbase + t0 + m0 + r + 8 * g) * (size_t)HID + h * KK_ + col] = acc[r];
  }
}

// ---------------- linear-attention scan (hyb += glob) ----------------------
// One workgroup per (b,h); S[128][128] f32 state in LDS; sequential 64 blocks.

__global__ __launch_bounds__(256) void linear_scan_kernel(
    const float* __restrict__ qkv, const float* __restrict__ phi_scale,
    const float* __restrict__ phi_bias, float* __restrict__ hyb) {
  __shared__ __align__(16) float lds[41152];       // ~161 KB (<320 KB/WGP)
  float* S    = lds;                // [128][128]
  float* pkT  = lds + 16384;        // [128][64] phi_k transposed; dead after Z
  float* vS   = lds + 24576;        // [64][128]
  float* pqS  = lds + 32768;        // [64][128]
  float* Zs   = lds + 40960;        // [128]
  float* den  = lds + 41088;        // [64]
  float* numS = pkT;                // alias: [64][128]

  int tid = threadIdx.x, lane = tid & 31, wave = tid >> 5;
  int bh = blockIdx.x, b = bh >> 4, h = bh & 15;
  size_t rowbase = (size_t)b * TT;

  for (int i = tid; i < 16384; i += 256) S[i] = 0.f;
  if (tid < 128) Zs[tid] = 0.f;
  __syncthreads();

  for (int nbk = 0; nbk < NB; ++nbk) {
    int t0 = nbk * BS;
    for (int i = tid; i < BS * KK_; i += 256) {
      int t = i >> 7, k = i & 127;
      size_t base = (rowbase + t0 + t) * (size_t)QKVN + h * KK_ + k;
      float ps = phi_scale[h * KK_ + k], pb = phi_bias[h * KK_ + k];
      pkT[k * BS + t] = phi_f(qkv[base + 2048], ps, pb);
      vS[i]  = qkv[base + 4096];
      pqS[i] = phi_f(qkv[base], ps, pb);
    }
    __syncthreads();

    // S += phi_k^T @ v : 128x128, contract 64 (A = pkT rows are k-dim)
    for (int tile = wave; tile < 64; tile += 8) {
      int m0 = (tile >> 3) * 16, n0 = (tile & 7) * 16;
      v8f acc = load_cd(S, KK_, m0, n0, lane);
      for (int kk = 0; kk < BS; kk += 4) {
        v2f a = load_f32_a(pkT, BS, m0, kk, lane);
        v2f bb = load_f32_b(vS, KK_, n0, kk, lane);
        acc = __builtin_amdgcn_wmma_f32_16x16x4_f32(false, a, false, bb,
                                                    (short)0, acc, false, false);
      }
      store_cd(S, KK_, m0, n0, lane, acc);
    }
    __syncthreads();

    if (tid < 128) { // Z += colsum(phi_k)
      float z = Zs[tid];
      for (int t = 0; t < BS; ++t) z += pkT[tid * BS + t];
      Zs[tid] = z;
    }
    __syncthreads();

    // num = phi_q @ S : 64x128, contract 128
    for (int tile = wave; tile < 32; tile += 8) {
      int m0 = (tile >> 3) * 16, n0 = (tile & 7) * 16;
      v8f acc = {};
      for (int kk = 0; kk < KK_; kk += 4) {
        v2f a = load_f32_a(pqS, KK_, m0, kk, lane);
        v2f bb = load_f32_b(S, KK_, n0, kk, lane);
        acc = __builtin_amdgcn_wmma_f32_16x16x4_f32(false, a, false, bb,
                                                    (short)0, acc, false, false);
      }
      store_cd(numS, KK_, m0, n0, lane, acc);
    }
    __syncthreads();

    if (tid < BS) { // den = max(phi_q . Z, eps)
      float d = 0.f;
      for (int k = 0; k < KK_; ++k) d += pqS[tid * KK_ + k] * Zs[k];
      den[tid] = fmaxf(d, EPS_);
    }
    __syncthreads();

    for (int i = tid; i < BS * KK_; i += 256) {
      int t = i >> 7, vd = i & 127;
      hyb[(rowbase + t0 + t) * (size_t)HID + h * KK_ + vd] += numS[i] / den[t];
    }
    __syncthreads();
  }
}

// ---------------- host launcher --------------------------------------------

extern "C" void kernel_launch(void* const* d_in, const int* in_sizes, int n_in,
                              void* d_out, int out_size, void* d_ws, size_t ws_size,
                              hipStream_t stream) {
  (void)in_sizes; (void)n_in; (void)out_size; (void)ws_size;
  const float* x   = (const float*)d_in[0];
  const float* Wq  = (const float*)d_in[1];
  const float* Wk  = (const float*)d_in[2];
  const float* Wv  = (const float*)d_in[3];
  const float* ps  = (const float*)d_in[4];
  const float* pb  = (const float*)d_in[5];
  const float* Wo  = (const float*)d_in[6];
  const float* bo  = (const float*)d_in[7];
  float* out = (float*)d_out;

  // workspace layout (~370 MB)
  char* p = (char*)d_ws;
  __bf16* xb    = (__bf16*)p; p += (size_t)BB * TT * HID * 2;   // 32 MB
  __bf16* wqkvT = (__bf16*)p; p += (size_t)HID * QKVN * 2;      // 24 MB  [6144][2048]
  __bf16* woT   = (__bf16*)p; p += (size_t)HID * HID * 2;       //  8 MB  [2048][2048]
  float*  qkv   = (float*)p;  p += (size_t)BB * TT * QKVN * 4;  // 192 MB
  float*  hyb   = (float*)p;  p += (size_t)BB * TT * HID * 4;   // 64 MB
  __bf16* hybb  = (__bf16*)p; p += (size_t)BB * TT * HID * 2;   // 32 MB

  const size_t nX = (size_t)BB * TT * HID;        // 16.7M
  const size_t nW = (size_t)HID * HID;            // 4.2M

  // 1. f32 -> bf16 conversions; weights transposed to [N][Kd]
  cvt_f32_bf16<<<(nX + 255) / 256, 256, 0, stream>>>(x, xb, nX);
  pack_wT_bf16<<<(nW + 255) / 256, 256, 0, stream>>>(Wq, wqkvT, HID, HID, 0);
  pack_wT_bf16<<<(nW + 255) / 256, 256, 0, stream>>>(Wk, wqkvT, HID, HID, 2048);
  pack_wT_bf16<<<(nW + 255) / 256, 256, 0, stream>>>(Wv, wqkvT, HID, HID, 4096);
  pack_wT_bf16<<<(nW + 255) / 256, 256, 0, stream>>>(Wo, woT, HID, HID, 0);

  // 2. fused QKV projection: [8192,2048] x [2048,6144]
  gemm_bf16_rcr<<<dim3(QKVN / GTN, (BB * TT) / GTM), 256, 0, stream>>>(
      xb, wqkvT, nullptr, qkv, BB * TT, QKVN, HID);

  // 3. block-local softmax attention -> hyb
  local_attn_kernel<<<BB * HH * NB, 256, 0, stream>>>(qkv, hyb);

  // 4. linear-attention scan, hyb += glob
  linear_scan_kernel<<<BB * HH, 256, 0, stream>>>(qkv, ps, pb, hyb);

  // 5. output projection: [8192,2048] x [2048,2048] + bo
  cvt_f32_bf16<<<(nX + 255) / 256, 256, 0, stream>>>(hyb, hybb, nX);
  gemm_bf16_rcr<<<dim3(HID / GTN, (BB * TT) / GTM), 256, 0, stream>>>(
      hybb, woT, bo, out, BB * TT, HID, HID);
}
